// DMoNPoolingLayer_88218628260320
// MI455X (gfx1250) — matrix operand order repaired and checked by reading, not compiled, not run on hardware
//
#include <hip/hip_runtime.h>

// ---------------------------------------------------------------------------
// Types for CDNA5 WMMA
// ---------------------------------------------------------------------------
typedef __attribute__((ext_vector_type(16))) __bf16 v16bf;
typedef __attribute__((ext_vector_type(8)))  float  v8f;

// Problem sizes (fixed by the reference)
#define BB 8
#define NN 4096
#define CC 128
#define KK 64

#define PITCH 40  // LDS row pitch in halfwords: 80 B, 16B-aligned rows, b128-conflict-free

// f32 -> bf16 round-to-nearest-even
__device__ __forceinline__ unsigned short f2bf(float f) {
    union { float f; unsigned u; } c;
    c.f = f;
    unsigned u = c.u;
    u += 0x7fffu + ((u >> 16) & 1u);
    return (unsigned short)(u >> 16);
}

__device__ __forceinline__ unsigned hw4(uint2 g, int idx) {
    unsigned w = (idx & 2) ? g.y : g.x;
    return (idx & 1) ? (w >> 16) : (w & 0xffffu);
}

// ---------------------------------------------------------------------------
// Fragment loaders: 2x ds_load_b128 each, no packing.
// ldsA is row-major [M(=k) 0..63][K(=inner) 0..31], pitch PITCH halfwords.
//   A frag k order used here: lane<16: M=lane,    K = 0..15 (contiguous)
//                          lane>=16: M=lane-16, K = 16..31 (contiguous)
// ldsBT is row-major [N(=out col) ][K(=inner) 0..31], pitch PITCH.
//   B frag k order used here: lane<16: col=lane,    K = 0..15 (contiguous)
//                          lane>=16: col=lane-16, K = 16..31 (contiguous)
// ---------------------------------------------------------------------------
__device__ __forceinline__ v16bf load_a_frag(const unsigned short* lds, int ktile, int lane) {
    const int row = ktile * 16 + (lane & 15);
    const int q0  = (lane < 16) ? 0 : 2;  // uint4 index: cols 0..15 or 16..31
    union { v16bf v; uint4 q[2]; } r;
    const uint4* p = (const uint4*)(lds + row * PITCH);
    r.q[0] = p[q0];
    r.q[1] = p[q0 + 1];   // cols +8
    return r.v;
}

__device__ __forceinline__ v16bf load_b_frag(const unsigned short* lds, int colbase, int lane) {
    const int row = colbase + (lane & 15);
    const int q0  = (lane < 16) ? 0 : 2;  // K 0..15 or 16..31
    union { v16bf v; uint4 q[2]; } r;
    const uint4* p = (const uint4*)(lds + row * PITCH);
    r.q[0] = p[q0];
    r.q[1] = p[q0 + 1];
    return r.v;
}

// ---------------------------------------------------------------------------
// Kernel 1: s = softmax(x @ W + b); emit f32 s [b][n][k], bf16 s TRANSPOSED
// [b][k][n], and cluster_size partials.
// Grid: (B*N)/32 blocks x 256 threads. 8 threads per row, 8 logits each.
// ---------------------------------------------------------------------------
__global__ void k1_assign(const float* __restrict__ x, const float* __restrict__ W,
                          const float* __restrict__ bias, float* __restrict__ s,
                          unsigned short* __restrict__ sbfT, float* __restrict__ csize) {
    __shared__ float sW[CC * KK];    // 32 KB
    __shared__ float sX[32 * CC];    // 16 KB
    __shared__ float red[256];
    __shared__ float red2[256];
    __shared__ float scs[KK];
    __shared__ float sbias[KK];
    __shared__ __align__(16) unsigned short stile[KK * 32];  // s bf16 [k][n-local]

    const int tid = threadIdx.x;
    const size_t base_row = (size_t)blockIdx.x * 32;
    const int b = (int)(base_row >> 12);       // rows within a block share one batch
    const int nloc0 = (int)(base_row & (NN - 1));

    for (int i = tid; i < CC * KK; i += 256) sW[i] = W[i];
    for (int i = tid; i < 32 * CC; i += 256) sX[i] = x[base_row * CC + i];
    if (tid < KK) { scs[tid] = 0.f; sbias[tid] = bias[tid]; }
    __syncthreads();

    const int r = tid >> 3;   // row 0..31
    const int q = tid & 7;    // logit group 0..7 (8 logits each)

    float lg[8];
    float mx = -3.4e38f;
#pragma unroll
    for (int kk = 0; kk < 8; ++kk) {
        int k = q * 8 + kk;
        float acc = sbias[k];
#pragma unroll 4
        for (int c = 0; c < CC; ++c) acc += sX[r * CC + c] * sW[c * KK + k];
        lg[kk] = acc;
        mx = fmaxf(mx, acc);
    }
    red[r * 8 + q] = mx;
    __syncthreads();
    float rmax = red[r * 8];
#pragma unroll
    for (int j = 1; j < 8; ++j) rmax = fmaxf(rmax, red[r * 8 + j]);

    float se = 0.f;
#pragma unroll
    for (int kk = 0; kk < 8; ++kk) { lg[kk] = __expf(lg[kk] - rmax); se += lg[kk]; }
    red2[r * 8 + q] = se;
    __syncthreads();
    float rsum = 0.f;
#pragma unroll
    for (int j = 0; j < 8; ++j) rsum += red2[r * 8 + j];
    float inv = 1.f / rsum;

#pragma unroll
    for (int kk = 0; kk < 8; ++kk) {
        int k = q * 8 + kk;
        float v = lg[kk] * inv;
        s[(base_row + (size_t)r) * KK + k] = v;
        stile[k * 32 + r] = f2bf(v);
        atomicAdd(&scs[k], v);
    }
    __syncthreads();
    if (tid < KK) atomicAdd(&csize[b * KK + tid], scs[tid]);

    // write bf16 s transposed: sbfT[b][k][n]  (16B per thread)
    {
        int k = tid >> 2, nseg = (tid & 3) * 8;
        const uint4* sv = (const uint4*)(stile + k * 32 + nseg);
        uint4* dv = (uint4*)(sbfT + ((size_t)b * KK + k) * NN + nloc0 + nseg);
        *dv = *sv;
    }
}

// ---------------------------------------------------------------------------
// Kernel 2: sT_adj[b,k,m] = sum_n s[b,n,k]*adj[b,n,m]  (stored [b][m][k], bf16)
//           plus deg[b,n] = adj row sums (atomic partials).
// Grid: B*32 blocks (128-col slabs) x 256 threads (8 waves).
// Single adj pass: 32x128 f32 chunks -> 4x4 register transpose -> bf16 LDS
// [m][n]; A staged as b128 copies of pre-transposed sbfT. All fragment loads
// are 2x ds_load_b128.
// ---------------------------------------------------------------------------
__global__ void k2_stadj(const float* __restrict__ adj, const unsigned short* __restrict__ sbfT,
                         unsigned short* __restrict__ sTadjT, float* __restrict__ deg) {
    const int b  = blockIdx.x >> 5;
    const int m0 = (blockIdx.x & 31) * 128;
    const float* adjB = adj + (size_t)b * NN * NN;

    __shared__ __align__(16) unsigned short ldsBT[128 * PITCH];  // adj^T chunk [m][n]
    __shared__ __align__(16) unsigned short ldsA[KK * PITCH];    // s^T chunk [k][n]
    __shared__ float sdeg[32];

    const int tid  = threadIdx.x;
    const int lane = tid & 31;
    const int wave = tid >> 5;
    const int mblk = tid & 31;   // 4-col block within 128
    const int nblk = tid >> 5;   // 4-row block within 32

    v8f acc[4];
#pragma unroll
    for (int t = 0; t < 4; ++t)
#pragma unroll
        for (int e = 0; e < 8; ++e) acc[t][e] = 0.f;

    for (int n0 = 0; n0 < NN; n0 += 32) {
        if (tid < 32) sdeg[tid] = 0.f;
        __syncthreads();  // also guards LDS reuse from previous iteration

        // stage adj 4x4 block: coalesced b128 loads, transpose, bf16-pack, b64 stores
        {
            const float* src = adjB + (size_t)(n0 + nblk * 4) * NN + m0 + mblk * 4;
            union { float4 f4; float f[4]; } v[4];
#pragma unroll
            for (int i = 0; i < 4; ++i) v[i].f4 = *(const float4*)(src + (size_t)i * NN);
            if (n0 + 32 < NN && (mblk & 7) == 0) {  // one prefetch per 128B line
#pragma unroll
                for (int i = 0; i < 4; ++i)
                    __builtin_prefetch(src + (size_t)(32 + i) * NN, 0, 1);
            }
#pragma unroll
            for (int i = 0; i < 4; ++i)
                atomicAdd(&sdeg[nblk * 4 + i], v[i].f[0] + v[i].f[1] + v[i].f[2] + v[i].f[3]);
#pragma unroll
            for (int j = 0; j < 4; ++j) {
                uint2 p;
                p.x = (unsigned)f2bf(v[0].f[j]) | ((unsigned)f2bf(v[1].f[j]) << 16);
                p.y = (unsigned)f2bf(v[2].f[j]) | ((unsigned)f2bf(v[3].f[j]) << 16);
                *(uint2*)(ldsBT + (mblk * 4 + j) * PITCH + nblk * 4) = p;
            }
        }
        // stage s^T: straight b128 copy from sbfT[b][k][n0..n0+31]
        {
            int k = tid >> 2, nseg = (tid & 3) * 8;
            const uint4* sv = (const uint4*)(sbfT + ((size_t)b * KK + k) * NN + n0 + nseg);
            *(uint4*)(ldsA + k * PITCH + nseg) = *sv;
        }
        __syncthreads();
        if (tid < 32) atomicAdd(&deg[(size_t)b * NN + n0 + tid], sdeg[tid]);

        v16bf bfrag = load_b_frag(ldsBT, wave * 16, lane);
#pragma unroll
        for (int kt = 0; kt < 4; ++kt) {
            v16bf afrag = load_a_frag(ldsA, kt, lane);
            acc[kt] = __builtin_amdgcn_wmma_f32_16x16x32_bf16(
                false, afrag, false, bfrag, (short)0, acc[kt], false, false);
        }
    }

    // store bf16 [b][m][k]: packed uint4 (8 bf16) per accumulator tile
    const int colm = m0 + wave * 16 + (lane & 15);
    const int rb   = (lane < 16) ? 0 : 8;
    unsigned short* dst = sTadjT + (size_t)b * NN * KK + (size_t)colm * KK;
#pragma unroll
    for (int kt = 0; kt < 4; ++kt) {
        uint4 qd;
        qd.x = (unsigned)f2bf(acc[kt][0]) | ((unsigned)f2bf(acc[kt][1]) << 16);
        qd.y = (unsigned)f2bf(acc[kt][2]) | ((unsigned)f2bf(acc[kt][3]) << 16);
        qd.z = (unsigned)f2bf(acc[kt][4]) | ((unsigned)f2bf(acc[kt][5]) << 16);
        qd.w = (unsigned)f2bf(acc[kt][6]) | ((unsigned)f2bf(acc[kt][7]) << 16);
        *(uint4*)(dst + kt * 16 + rb) = qd;
    }
}

// ---------------------------------------------------------------------------
// Kernel 3: D[64 x 64-col-group] = A[64 x 4096] * B[4096 x cols]
// ATR=false: A given as [b][k][inner] bf16 (sbfT)          -> b128 copy stage
// ATR=true : A given as [b][inner][64] bf16 (sTadjT)       -> 4x4 bf16 transpose
// BF32=false: B given as [b][col][inner] bf16 (sbfT)       -> b128 copy stage
// BF32=true : B given as [b][inner][ldB] f32 (x)           -> 4x4 f32 transpose+cvt
// Grid: B*nColGroups blocks x 128 threads (4 waves, full M=64 each, 16 cols).
// ---------------------------------------------------------------------------
template <bool ATR, bool BF32>
__global__ void k3_gemm64(const unsigned short* __restrict__ A, const void* __restrict__ Bsrc,
                          float* __restrict__ D, int ldB, int nColGroups) {
    const int b  = blockIdx.x / nColGroups;
    const int c0 = (blockIdx.x % nColGroups) * 64;

    __shared__ __align__(16) unsigned short ldsA[KK * PITCH];
    __shared__ __align__(16) unsigned short ldsBT[KK * PITCH];

    const int tid = threadIdx.x, lane = tid & 31, wave = tid >> 5;

    v8f acc[4];
#pragma unroll
    for (int t = 0; t < 4; ++t)
#pragma unroll
        for (int e = 0; e < 8; ++e) acc[t][e] = 0.f;

    for (int n0 = 0; n0 < NN; n0 += 32) {
        __syncthreads();
        // ---- stage A ----
        if (!ATR) {
            int k = tid >> 1, nseg = (tid & 1) * 16;
            const uint4* sv = (const uint4*)(A + ((size_t)b * KK + k) * NN + n0 + nseg);
            uint4* dv = (uint4*)(ldsA + k * PITCH + nseg);
            dv[0] = sv[0];
            dv[1] = sv[1];
        } else {
            const unsigned short* Ab = A + (size_t)b * NN * KK;
            int kblk = tid & 15, mblk = tid >> 4;
            uint2 g[4];
#pragma unroll
            for (int i = 0; i < 4; ++i)
                g[i] = *(const uint2*)(Ab + (size_t)(n0 + mblk * 4 + i) * KK + kblk * 4);
#pragma unroll
            for (int kq = 0; kq < 4; ++kq) {
                uint2 p;
                p.x = hw4(g[0], kq) | (hw4(g[1], kq) << 16);
                p.y = hw4(g[2], kq) | (hw4(g[3], kq) << 16);
                *(uint2*)(ldsA + (kblk * 4 + kq) * PITCH + mblk * 4) = p;
            }
        }
        // ---- stage B (transposed layout [col][inner]) ----
        if (!BF32) {
            const unsigned short* Bh = (const unsigned short*)Bsrc + (size_t)b * KK * NN;
            int col = tid >> 1, nseg = (tid & 1) * 16;
            const uint4* sv = (const uint4*)(Bh + (size_t)(c0 + col) * NN + n0 + nseg);
            uint4* dv = (uint4*)(ldsBT + col * PITCH + nseg);
            dv[0] = sv[0];
            dv[1] = sv[1];
        } else {
            const float* Bf = (const float*)Bsrc + (size_t)b * NN * ldB;
            int cblk = tid & 15, nblk = tid >> 4;
            union { float4 f4; float f[4]; } v[4];
#pragma unroll
            for (int i = 0; i < 4; ++i)
                v[i].f4 = *(const float4*)(Bf + (size_t)(n0 + nblk * 4 + i) * ldB + c0 + cblk * 4);
#pragma unroll
            for (int j = 0; j < 4; ++j) {
                uint2 p;
                p.x = (unsigned)f2bf(v[0].f[j]) | ((unsigned)f2bf(v[1].f[j]) << 16);
                p.y = (unsigned)f2bf(v[2].f[j]) | ((unsigned)f2bf(v[3].f[j]) << 16);
                *(uint2*)(ldsBT + (cblk * 4 + j) * PITCH + nblk * 4) = p;
            }
        }
        __syncthreads();

        v16bf bfrag = load_b_frag(ldsBT, wave * 16, lane);
#pragma unroll
        for (int kt = 0; kt < 4; ++kt) {
            v16bf afrag = load_a_frag(ldsA, kt, lane);
            acc[kt] = __builtin_amdgcn_wmma_f32_16x16x32_bf16(
                false, afrag, false, bfrag, (short)0, acc[kt], false, false);
        }
    }

    const int col = c0 + wave * 16 + (lane & 15);
    const int rb  = (lane < 16) ? 0 : 8;
    float* dst = D + (size_t)b * KK * ldB + col;
#pragma unroll
    for (int kt = 0; kt < 4; ++kt)
#pragma unroll
        for (int r = 0; r < 8; ++r)
            dst[(size_t)(kt * 16 + rb + r) * ldB] = acc[kt][r];
}

// ---------------------------------------------------------------------------
// Kernel 4: per-batch epilogue: losses, SELU, out_adj normalization.
// Grid: B blocks x 256 threads.
// ---------------------------------------------------------------------------
__device__ __forceinline__ float blk_reduce256(float* sred, int tid, float v) {
    __syncthreads();
    sred[tid] = v;
    __syncthreads();
    for (int st = 128; st > 0; st >>= 1) {
        if (tid < st) sred[tid] += sred[tid + st];
        __syncthreads();
    }
    float r = sred[0];
    __syncthreads();
    return r;
}

__global__ void k4_final(const float* __restrict__ s, const float* __restrict__ deg,
                         const float* __restrict__ csize, const float* __restrict__ ss,
                         const float* __restrict__ oadjraw, const float* __restrict__ pool,
                         float* __restrict__ out, float* __restrict__ out_adj,
                         float* __restrict__ losses) {
    __shared__ float sred[256];
    __shared__ float sca[KK];
    __shared__ float sd[KK];

    const int b = blockIdx.x, tid = threadIdx.x;
    const float* degb = deg + (size_t)b * NN;

    // 2m = sum of degrees
    float p = 0.f;
    for (int n = tid; n < NN; n += 256) p += degb[n];
    const float twoM = blk_reduce256(sred, tid, p);

    // ca[k] = sum_n s[n,k]*deg[n]
    {
        int k = tid & 63, g = tid >> 6;
        const float* sb = s + (size_t)b * NN * KK;
        float a = 0.f;
        for (int n = g; n < NN; n += 4) a += sb[(size_t)n * KK + k] * degb[n];
        __syncthreads();
        sred[tid] = a;
        __syncthreads();
        if (tid < KK) sca[tid] = sred[tid] + sred[tid + 64] + sred[tid + 128] + sred[tid + 192];
        __syncthreads();
    }

    // spectral loss
    const float* oa = oadjraw + (size_t)b * KK * KK;
    float tr = 0.f, c2 = 0.f;
    if (tid < KK) { tr = oa[tid * KK + tid]; c2 = sca[tid] * sca[tid]; }
    const float trsum = blk_reduce256(sred, tid, tr);
    const float casq  = blk_reduce256(sred, tid, c2);
    if (tid == 0) atomicAdd(&losses[0], (-(trsum - casq / twoM) / twoM) * (1.0f / BB));

    // orthogonality loss
    const float* ssb = ss + (size_t)b * KK * KK;
    float f2 = 0.f;
    for (int i = tid; i < KK * KK; i += 256) { float v = ssb[i]; f2 += v * v; }
    const float fro = sqrtf(blk_reduce256(sred, tid, f2));
    float o2 = 0.f;
    for (int i = tid; i < KK * KK; i += 256) {
        int kk = i >> 6, ll = i & 63;
        float v = ssb[i] / fro - (kk == ll ? 0.125f : 0.f);  // 1/sqrt(64)
        o2 += v * v;
    }
    const float ortho = sqrtf(blk_reduce256(sred, tid, o2));
    if (tid == 0) atomicAdd(&losses[1], ortho * (1.0f / BB));

    // cluster-size loss
    float cs2 = (tid < KK) ? csize[b * KK + tid] * csize[b * KK + tid] : 0.f;
    const float cl = sqrtf(blk_reduce256(sred, tid, cs2)) * (8.0f / (float)NN) - 1.0f;
    if (tid == 0) atomicAdd(&losses[2], cl * (1.0f / BB));

    // out_adj: zero diagonal, symmetric degree-normalize
    {
        int k = tid & 63, g = tid >> 6;
        float rs = 0.f;
        for (int l = g; l < KK; l += 4)
            if (l != k) rs += oa[k * KK + l];
        __syncthreads();
        sred[tid] = rs;
        __syncthreads();
        if (tid < KK)
            sd[tid] = sqrtf(sred[tid] + sred[tid + 64] + sred[tid + 128] + sred[tid + 192]) + 1e-15f;
        __syncthreads();
    }
    float* oout = out_adj + (size_t)b * KK * KK;
    for (int i = tid; i < KK * KK; i += 256) {
        int kk = i >> 6, ll = i & 63;
        oout[i] = (kk == ll) ? 0.f : oa[i] / (sd[kk] * sd[ll]);
    }

    // out = selu(pooled)
    const float* pb = pool + (size_t)b * KK * CC;
    float* ob = out + (size_t)b * KK * CC;
    const float SC = 1.0507009873554805f, AL = 1.6732632423543772f;
    for (int i = tid; i < KK * CC; i += 256) {
        float v = pb[i];
        ob[i] = SC * (v > 0.f ? v : AL * (__expf(v) - 1.f));
    }
}

// ---------------------------------------------------------------------------
// Host launcher
// ---------------------------------------------------------------------------
extern "C" void kernel_launch(void* const* d_in, const int* in_sizes, int n_in,
                              void* d_out, int out_size, void* d_ws, size_t ws_size,
                              hipStream_t stream) {
    (void)in_sizes; (void)n_in; (void)out_size; (void)ws_size;

    const float* x    = (const float*)d_in[0];
    const float* adj  = (const float*)d_in[1];
    const float* W    = (const float*)d_in[2];
    const float* bias = (const float*)d_in[3];

    // d_out layout: s | out | out_adj | 3 loss scalars
    const size_t OUT_OFF  = (size_t)BB * NN * KK;              // 2097152
    const size_t ADJ_OFF  = OUT_OFF + (size_t)BB * KK * CC;    // 2162688
    const size_t LOSS_OFF = ADJ_OFF + (size_t)BB * KK * KK;    // 2195456
    float* s_out   = (float*)d_out;
    float* out_ptr = (float*)d_out + OUT_OFF;
    float* oadj_o  = (float*)d_out + ADJ_OFF;
    float* losses  = (float*)d_out + LOSS_OFF;

    // workspace layout (bytes)
    unsigned char* w = (unsigned char*)d_ws;
    unsigned short* sbfT   = (unsigned short*)(w + 0);         // B*K*N bf16  = 4 MB (transposed s)
    unsigned short* stadjT = (unsigned short*)(w + 4194304);   // B*N*K bf16  = 4 MB ([m][k])
    float* deg    = (float*)(w + 8388608);                     // B*N f32     = 128 KB
    float* csize  = (float*)(w + 8519680);                     // B*K f32     = 2 KB
    float* ssb    = (float*)(w + 8521728);                     // B*K*K f32   = 128 KB
    float* oadjr  = (float*)(w + 8652800);                     // B*K*K f32   = 128 KB
    float* pool   = (float*)(w + 8783872);                     // B*K*C f32   = 256 KB

    // zero accumulators (deg + csize are contiguous) and the loss scalars
    hipMemsetAsync(w + 8388608, 0, 131072 + 2048, stream);
    hipMemsetAsync(losses, 0, 3 * sizeof(float), stream);

    // 1) cluster assignment softmax (+ transposed bf16 copy of s)
    k1_assign<<<(BB * NN) / 32, 256, 0, stream>>>(x, W, bias, s_out, sbfT, csize);

    // 2) S^T A (single full pass over the 512 MB adj) + degrees
    k2_stadj<<<BB * 32, 256, 0, stream>>>(adj, sbfT, stadjT, deg);

    // 3) small 64-row WMMA GEMMs
    k3_gemm64<false, false><<<BB,     128, 0, stream>>>(sbfT,   sbfT, ssb,   64, 1); // S^T S
    k3_gemm64<false, true ><<<BB * 2, 128, 0, stream>>>(sbfT,   x,    pool, 128, 2); // S^T X
    k3_gemm64<true,  false><<<BB,     128, 0, stream>>>(stadjT, sbfT, oadjr, 64, 1); // (S^T A) S

    // 4) losses + SELU + normalized pooled adjacency
    k4_final<<<BB, 256, 0, stream>>>(s_out, deg, csize, ssb, oadjr, pool,
                                     out_ptr, oadj_o, losses);
}
